// DGCNNGeom_74680891343000
// MI455X (gfx1250) — matrix operand order repaired and checked by reading, not compiled, not run on hardware
//
#include <hip/hip_runtime.h>

typedef __attribute__((ext_vector_type(16))) __bf16 v16bf;
typedef __attribute__((ext_vector_type(8)))  float  v8f;

#define FOUT 64
#define LRELU_SLOPE 0.2f
#define BN_EPS 1e-5f

// ---- ordered-uint encoding for float max (lowers to global_atomic_max_u32) ----
__device__ __forceinline__ unsigned fkey(float x) {
    unsigned b = __float_as_uint(x);
    return (b & 0x80000000u) ? ~b : (b | 0x80000000u);
}
__device__ __forceinline__ float fdecode(unsigned k) {
    unsigned b = (k & 0x80000000u) ? (k ^ 0x80000000u) : ~k;
    return __uint_as_float(b);
}

// Store one even-aligned pair of K values of the A fragment (ISA 16-bit A 16x32 layout).
__device__ __forceinline__ void storeApair(__bf16* tile, int m, int kpos, float v0, float v1) {
    int ks   = kpos >> 5;
    int kk   = kpos & 31;
    int lane = m + ((kk & 8) ? 16 : 0);
    int ii   = ((kk & 16) ? 8 : 0) + (kk & 7);
    __bf16* p = tile + ((ks << 5) + lane) * 16 + ii;
    p[0] = (__bf16)v0;
    p[1] = (__bf16)v1;
}

// ============================================================================
// EdgeConv: m[e,:] = [x_dst | x_src - x_dst] @ W   (bias dropped: cancels in BN)
// STATS=true : accumulate per-feature sum / sumsq of m over all edges
// STATS=false: apply BN(scale/shift)+LeakyReLU, scatter-max into out[dst]
// ============================================================================
template<int FIN, bool STATS>
__global__ __launch_bounds__(256) void k_edgeconv(
    const float* __restrict__ xnode,
    const int*   __restrict__ src,
    const int*   __restrict__ dst,
    const float* __restrict__ W,        // [2*FIN, 64] row-major f32
    float*       __restrict__ stats,    // [256]: sum64|sumsq64|scale64|shift64
    unsigned*    __restrict__ out,      // [N,64] encoded keys (agg pass only)
    int E, int N)
{
    constexpr int KREAL  = 2 * FIN;
    constexpr int KSTEPS = (KREAL + 31) / 32;
    constexpr int KTOT   = KSTEPS * 32;

    __shared__ __align__(32) __bf16 sB[KSTEPS * 4 * 32 * 16]; // W fragments
    __shared__ __align__(32) __bf16 sA[8 * KSTEPS * 32 * 16]; // 8 tiles x 16 edges
    __shared__ float sStats[128];
    __shared__ int   sDst[128];

    const int tid   = threadIdx.x;
    const int eBase = blockIdx.x * 128;

    // ---- stage B (W -> bf16 fragments, ISA 16-bit B 32x16 layout) ----
    for (int idx = tid; idx < KTOT * 64; idx += 256) {
        int k = idx >> 6, n = idx & 63;
        float w = (k < KREAL) ? W[k * 64 + n] : 0.f;
        int f    = ((k >> 5) << 2) + (n >> 4);
        int kk   = k & 31, nn = n & 15;
        int lane = nn + ((kk & 8) ? 16 : 0);
        int ii   = ((kk & 16) ? 8 : 0) + (kk & 7);
        sB[(f * 32 + lane) * 16 + ii] = (__bf16)w;
    }
    if (STATS && tid < 128) sStats[tid] = 0.f;

    // ---- stage A (gather edge features, convert to bf16 fragments) ----
    if constexpr (FIN == 3) {
        for (int idx = tid; idx < (8 * KSTEPS * 32 * 16) / 2; idx += 256)
            ((unsigned*)sA)[idx] = 0u;                 // zero K-padding
        __syncthreads();
        if (tid < 128) {
            int e = eBase + tid;
            if (e < E) {
                int s0 = src[e], d0 = dst[e];
                sDst[tid] = d0;
                const float* xi = xnode + (size_t)d0 * 3;
                const float* xj = xnode + (size_t)s0 * 3;
                float a0 = xi[0], a1 = xi[1], a2 = xi[2];
                float b0 = xj[0] - a0, b1 = xj[1] - a1, b2 = xj[2] - a2;
                __bf16* tileA = sA + (tid >> 4) * (KSTEPS * 512);
                int m = tid & 15;
                storeApair(tileA, m, 0, a0, a1);
                storeApair(tileA, m, 2, a2, b0);
                storeApair(tileA, m, 4, b1, b2);
            } else {
                sDst[tid] = 0;
            }
        }
    } else {
        int eLocal = tid >> 1;
        int h      = tid & 1;
        int e      = eBase + eLocal;
        int m      = eLocal & 15;
        int kbase  = h * 32;
        __bf16* tileA = sA + (eLocal >> 4) * (KSTEPS * 512);
        if (e < E) {
            int s0 = src[e], d0 = dst[e];
            if (h == 0) sDst[eLocal] = d0;
            const float* xi = xnode + (size_t)d0 * 64 + kbase;
            const float* xj = xnode + (size_t)s0 * 64 + kbase;
            #pragma unroll
            for (int q = 0; q < 8; ++q) {
                float4 a = *(const float4*)(xi + q * 4);
                float4 b = *(const float4*)(xj + q * 4);
                storeApair(tileA, m, kbase + q * 4,          a.x,        a.y);
                storeApair(tileA, m, kbase + q * 4 + 2,      a.z,        a.w);
                storeApair(tileA, m, 64 + kbase + q * 4,     b.x - a.x,  b.y - a.y);
                storeApair(tileA, m, 64 + kbase + q * 4 + 2, b.z - a.z,  b.w - a.w);
            }
        } else {
            if (h == 0) sDst[eLocal] = 0;
            #pragma unroll
            for (int q = 0; q < 8; ++q) {
                storeApair(tileA, m, kbase + q * 4,          0.f, 0.f);
                storeApair(tileA, m, kbase + q * 4 + 2,      0.f, 0.f);
                storeApair(tileA, m, 64 + kbase + q * 4,     0.f, 0.f);
                storeApair(tileA, m, 64 + kbase + q * 4 + 2, 0.f, 0.f);
            }
        }
    }
    __syncthreads();

    // ---- WMMA: 16 edges x 64 outputs per wave ----
    const int wid  = tid >> 5;
    const int lane = tid & 31;
    const __bf16* tA = sA + wid * (KSTEPS * 512);

    v8f acc[4] = {};
    #pragma unroll
    for (int ks = 0; ks < KSTEPS; ++ks) {
        v16bf af = *(const v16bf*)(tA + (ks * 32 + lane) * 16);
        #pragma unroll
        for (int nt = 0; nt < 4; ++nt) {
            v16bf bf = *(const v16bf*)(sB + ((ks * 4 + nt) * 32 + lane) * 16);
            acc[nt] = __builtin_amdgcn_wmma_f32_16x16x32_bf16(
                false, af, false, bf, (short)0, acc[nt], false, false);
        }
    }

    if constexpr (STATS) {
        // C layout: lane L and L^16 hold the same column n = nt*16 + (L&15)
        #pragma unroll
        for (int nt = 0; nt < 4; ++nt) {
            float s = 0.f, q = 0.f;
            #pragma unroll
            for (int r = 0; r < 8; ++r) { float v = acc[nt][r]; s += v; q += v * v; }
            s += __shfl_xor(s, 16, 32);
            q += __shfl_xor(q, 16, 32);
            if (lane < 16) {
                int n = nt * 16 + lane;
                atomicAdd(&sStats[n], s);
                atomicAdd(&sStats[64 + n], q);
            }
        }
        __syncthreads();
        if (tid < 128) atomicAdd(&stats[tid], sStats[tid]);
    } else {
        #pragma unroll
        for (int nt = 0; nt < 4; ++nt) {
            int n = nt * 16 + (lane & 15);
            float sc = stats[128 + n];
            float sh = stats[192 + n];
            #pragma unroll
            for (int r = 0; r < 8; ++r) {
                int m = r + ((lane & 16) ? 8 : 0);
                int e = eBase + wid * 16 + m;
                if (e < E) {
                    float v = sc * acc[nt][r] + sh;
                    v = v > 0.f ? v : LRELU_SLOPE * v;
                    int d0 = sDst[wid * 16 + m];
                    atomicMax(&out[(size_t)d0 * 64 + n], fkey(v));
                }
            }
        }
    }
}

// ---- helpers ----
__global__ void k_init(float* stats, unsigned* agg, int nAgg) {
    int i = blockIdx.x * blockDim.x + threadIdx.x;
    if (i < 256) stats[i] = 0.f;
    if (i < nAgg) agg[i] = 0u;
}

__global__ void k_zero_stats(float* stats) {
    int i = threadIdx.x;
    if (i < 256) stats[i] = 0.f;
}

__global__ void k_bn_prep(float* stats, const float* __restrict__ gamma,
                          const float* __restrict__ beta, float invCount, int nf) {
    int f = blockIdx.x * blockDim.x + threadIdx.x;
    if (f >= nf) return;
    float mean = stats[f] * invCount;
    float var  = stats[nf + f] * invCount - mean * mean;
    float sc   = gamma[f] * rsqrtf(var + BN_EPS);
    stats[2 * nf + f] = sc;
    stats[3 * nf + f] = beta[f] - mean * sc;
}

__global__ void k_fixup(unsigned* buf, int n) {
    int i = blockIdx.x * blockDim.x + threadIdx.x;
    if (i >= n) return;
    unsigned k = buf[i];
    float v = (k == 0u) ? 0.f : fdecode(k);
    if (!__builtin_isfinite(v)) v = 0.f;
    ((float*)buf)[i] = v;
}

// ---- output head: o1 = h @ Wo1 (bias cancels in BN), plus BN stats over N ----
__global__ __launch_bounds__(256) void k_out1(const float* __restrict__ h,
                                              const float* __restrict__ Wo1,
                                              float* __restrict__ o1,
                                              float* __restrict__ stats, int n) {
    __shared__ float sW[64 * 16];
    __shared__ float sSt[32];
    int tid = threadIdx.x;
    for (int i = tid; i < 64 * 16; i += 256) sW[i] = Wo1[i];
    if (tid < 32) sSt[tid] = 0.f;
    __syncthreads();
    int node = blockIdx.x * 256 + tid;
    float a[16];
    #pragma unroll
    for (int f = 0; f < 16; ++f) a[f] = 0.f;
    if (node < n) {
        const float* row = h + (size_t)node * 64;
        #pragma unroll 8
        for (int k = 0; k < 64; ++k) {
            float xv = row[k];
            #pragma unroll
            for (int f = 0; f < 16; ++f) a[f] += xv * sW[k * 16 + f];
        }
        #pragma unroll
        for (int f = 0; f < 16; ++f) o1[(size_t)node * 16 + f] = a[f];
    }
    #pragma unroll
    for (int f = 0; f < 16; ++f) {
        atomicAdd(&sSt[f], a[f]);
        atomicAdd(&sSt[16 + f], a[f] * a[f]);
    }
    __syncthreads();
    if (tid < 32) atomicAdd(&stats[tid], sSt[tid]);
}

__global__ __launch_bounds__(256) void k_out2(const float* __restrict__ o1,
                                              const float* __restrict__ stats,
                                              const float* __restrict__ Wo2,
                                              const float* __restrict__ bo2,
                                              float* __restrict__ out, int n) {
    __shared__ float sW[16 * 40];
    __shared__ float sb[40];
    __shared__ float sSc[16], sSh[16];
    int tid = threadIdx.x;
    for (int i = tid; i < 640; i += 256) sW[i] = Wo2[i];
    if (tid < 40) sb[tid] = bo2[tid];
    if (tid < 16) { sSc[tid] = stats[32 + tid]; sSh[tid] = stats[48 + tid]; }
    __syncthreads();
    int node = blockIdx.x * 256 + tid;
    if (node >= n) return;
    float a[16];
    #pragma unroll
    for (int f = 0; f < 16; ++f) {
        float v = sSc[f] * o1[(size_t)node * 16 + f] + sSh[f];
        a[f] = v > 0.f ? v : LRELU_SLOPE * v;
    }
    #pragma unroll 4
    for (int c = 0; c < 40; ++c) {
        float acc = sb[c];
        #pragma unroll
        for (int f = 0; f < 16; ++f) acc += a[f] * sW[f * 40 + c];
        out[(size_t)node * 40 + c] = acc;
    }
}

extern "C" void kernel_launch(void* const* d_in, const int* in_sizes, int n_in,
                              void* d_out, int out_size, void* d_ws, size_t ws_size,
                              hipStream_t stream) {
    const float* x    = (const float*)d_in[0];
    const int*   ei   = (const int*)d_in[1];
    const float* W0   = (const float*)d_in[2];   // [6,64]
    const float* g0   = (const float*)d_in[4];
    const float* be0  = (const float*)d_in[5];
    const float* Ws   = (const float*)d_in[6];   // [3,128,64]
    const float* gs   = (const float*)d_in[8];
    const float* bes  = (const float*)d_in[9];
    const float* Wo1  = (const float*)d_in[10];  // [64,16]
    const float* go   = (const float*)d_in[12];
    const float* beo  = (const float*)d_in[13];
    const float* Wo2  = (const float*)d_in[14];  // [16,40]
    const float* bo2  = (const float*)d_in[15];

    const int N = in_sizes[0] / 3;
    const int E = in_sizes[1] / 2;
    const int* src = ei;
    const int* dst = ei + E;

    float*    stats = (float*)d_ws;              // 256 f32
    float*    hA    = stats + 256;               // N*64 f32
    float*    hB    = hA + (size_t)N * 64;       // N*64 f32

    const int gridE    = (E + 127) / 128;
    const int gridNF   = (N * 64 + 255) / 256;
    const int gridNode = (N + 255) / 256;
    const float invE   = 1.0f / (float)E;
    const float invN   = 1.0f / (float)N;

    // ---- layer 0: x (FIN=3) -> hA ----
    k_init<<<gridNF, 256, 0, stream>>>(stats, (unsigned*)hA, N * 64);
    k_edgeconv<3, true ><<<gridE, 256, 0, stream>>>(x, src, dst, W0, stats, nullptr, E, N);
    k_bn_prep<<<1, 64, 0, stream>>>(stats, g0, be0, invE, 64);
    k_edgeconv<3, false><<<gridE, 256, 0, stream>>>(x, src, dst, W0, stats, (unsigned*)hA, E, N);
    k_fixup<<<gridNF, 256, 0, stream>>>((unsigned*)hA, N * 64);

    // ---- layers 1..3: FIN=64 ping-pong ----
    float* cur = hA;
    float* nxt = hB;
    for (int i = 0; i < 3; ++i) {
        const float* W  = Ws  + (size_t)i * 128 * 64;
        const float* ga = gs  + i * 64;
        const float* be = bes + i * 64;
        k_init<<<gridNF, 256, 0, stream>>>(stats, (unsigned*)nxt, N * 64);
        k_edgeconv<64, true ><<<gridE, 256, 0, stream>>>(cur, src, dst, W, stats, nullptr, E, N);
        k_bn_prep<<<1, 64, 0, stream>>>(stats, ga, be, invE, 64);
        k_edgeconv<64, false><<<gridE, 256, 0, stream>>>(cur, src, dst, W, stats, (unsigned*)nxt, E, N);
        k_fixup<<<gridNF, 256, 0, stream>>>((unsigned*)nxt, N * 64);
        float* t = cur; cur = nxt; nxt = t;
    }

    // ---- output head: cur -> o1 (reuse free ping-pong buffer) -> d_out ----
    float* o1 = nxt;
    k_zero_stats<<<1, 256, 0, stream>>>(stats);
    k_out1<<<gridNode, 256, 0, stream>>>(cur, Wo1, o1, stats, N);
    k_bn_prep<<<1, 64, 0, stream>>>(stats, go, beo, invN, 16);
    k_out2<<<gridNode, 256, 0, stream>>>(o1, stats, Wo2, bo2, (float*)d_out, N);
}
